// Algin_45672682226251
// MI455X (gfx1250) — matrix operand order repaired and checked by reading, not compiled, not run on hardware
//
#include <hip/hip_runtime.h>

// ---------------------------------------------------------------------------
// CDNA5 (gfx1250) deformable-alignment network.
// NHWC f16 activations; every conv / deform-conv is an implicit GEMM on
// v_wmma_f32_16x16x32_f16 with K ordered (tap r)*CPAD + cin, so each lane's
// 16-K run is one (tap, 16-channel block) -> one aligned 32B LDS load.
// Conv kernels are templated on <CPAD, TS>: the K loop unrolls into
// straight-line  global_b128(A) + ds_b128(B, imm offset) + v_wmma  code.
// Deform conv uses clamped unconditional corner loads with validity folded
// into the bilinear weights (no EXEC divergence), packed f16 combine.
// Halo windows are staged into LDS with global_load_async_to_lds_b128.
// ---------------------------------------------------------------------------

typedef __attribute__((ext_vector_type(16))) _Float16 v16h;
typedef __attribute__((ext_vector_type(8)))  _Float16 v8h;
typedef __attribute__((ext_vector_type(8)))  float    v8f;
typedef __attribute__((ext_vector_type(4)))  int      v4i;

#ifndef HAVE_ASYNC_LDS
#if defined(__has_builtin)
#if __has_builtin(__builtin_amdgcn_global_load_async_to_lds_b128)
#define HAVE_ASYNC_LDS 1
#else
#define HAVE_ASYNC_LDS 0
#endif
#else
#define HAVE_ASYNC_LDS 0
#endif
#endif

#define AS1 __attribute__((address_space(1)))
#define AS3 __attribute__((address_space(3)))

union U16h { v16h v; _Float16 h[16]; };
union U8h  { v8h  v; _Float16 h[8];  };
union U8f  { v8f  v; float    f[8];  };

// --------------------------- fp32 -> fp16 copy -----------------------------
__global__ __launch_bounds__(256)
void f32_to_f16_kernel(const float* __restrict__ in, _Float16* __restrict__ out,
                       long total)
{
    long i = (long)blockIdx.x * 256 + threadIdx.x;
    if (i < total) out[i] = (_Float16)in[i];
}

// ---------------------------------------------------------------------------
// Pack OIHW fp32 weights into f16 WMMA A-fragments with K = r*Cpad + cin.
// pw[(((mtile*Ksteps + ks)*32 + lane)*16 + j)], j = vgpr*2 + half.
// Packed A 16x32 layout (contiguous K per lane half): lane L -> M = L&15; kbase = (L>>4)*8;
//   V0..V7 hold K = 2*kbase+{0..15}, i.e. K = 16*(L>>4) + j.
// Pad region (r>=9 or cin>=Cin or m>=Cout) is zero, which is what lets the
// compute kernels skip zero-guards on the B side.
// ---------------------------------------------------------------------------
__global__ __launch_bounds__(256)
void pack_w_kernel(const float* __restrict__ w, _Float16* __restrict__ pw,
                   int Cout, int Cin, int Cpad, int Ksteps, int Mtiles)
{
    long total = (long)Mtiles * Ksteps * 512;
    long idx = (long)blockIdx.x * 256 + threadIdx.x;
    if (idx >= total) return;
    int j    = (int)(idx & 15);
    int lane = (int)((idx >> 4) & 31);
    long rest = idx >> 9;
    int ks = (int)(rest % Ksteps);
    int mt = (int)(rest / Ksteps);
    int v = j >> 1, hh = j & 1;
    int kbase = (lane >> 4) * 8;
    int kl = kbase * 2 + v * 2 + hh;
    int k = ks * 32 + kl;
    int m = mt * 16 + (lane & 15);
    int r = k / Cpad, cin = k - r * Cpad;     // K order: tap-major, channel-minor
    float val = 0.f;
    if (r < 9 && cin < Cin && m < Cout)
        val = w[(size_t)(m * Cin + cin) * 9 + r];   // OIHW, 3x3 contiguous
    pw[idx] = (_Float16)val;
}

// ---------------------------------------------------------------------------
// 3x3 conv (stride 1, pad 1) on up-to-two concatenated NHWC-f16 sources.
// One block = one output-row strip of TS pixels ((TS/16) waves), one mtile,
// one batch.  3 x (TS+2) x CPAD halo window in LDS.
// ---------------------------------------------------------------------------
template<int CPAD, int TS>
__global__ __launch_bounds__(256)
void conv3x3_nhwc_wmma_kernel(
    const _Float16* __restrict__ src0, int C0,
    const _Float16* __restrict__ src1, int C1,
    int H, int W,
    const v16h* __restrict__ gA,
    const float* __restrict__ bias, int Cout, int relu,
    _Float16* __restrict__ out)
{
    constexpr int KSTEPS = (9 * CPAD + 31) / 32;
    constexpr int WIN    = TS + 2;
    constexpr int CB     = CPAD / 16;     // channel blocks
    constexpr int GV     = 9 * CB;        // valid 16-K groups
    __shared__ __align__(32) _Float16 sact[3 * WIN * CPAD];

    const int lane  = threadIdx.x & 31;
    const int wave  = threadIdx.x >> 5;
    const int mtile = blockIdx.y;
    const int n     = blockIdx.z;
    const int spr   = W / TS;
    const int y0    = blockIdx.x / spr;
    const int xbase = (blockIdx.x - y0 * spr) * TS;
    const size_t hw = (size_t)H * W;
    const int Ct    = C0 + C1;

    // ---- stage halo window (zero padded to CPAD channels) ----
    if (CPAD == 16 && C0 == 16 && C1 == 0) {
        for (int t = threadIdx.x; t < 3 * WIN; t += blockDim.x) {
            int ry = t / WIN, xx = t - ry * WIN;
            int y = y0 - 1 + ry, x = xbase - 1 + xx;
            v8h* d = (v8h*)&sact[t * CPAD];
            if (y >= 0 && y < H && x >= 0 && x < W) {
                v8h* s = (v8h*)(src0 + ((size_t)n * hw + (size_t)y * W + x) * 16);
#if HAVE_ASYNC_LDS
                __builtin_amdgcn_global_load_async_to_lds_b128(
                    (AS1 v4i*)(s),     (AS3 v4i*)(d),     0, 0);
                __builtin_amdgcn_global_load_async_to_lds_b128(
                    (AS1 v4i*)(s + 1), (AS3 v4i*)(d + 1), 0, 0);
#else
                d[0] = s[0]; d[1] = s[1];
#endif
            } else { v8h z = {}; d[0] = z; d[1] = z; }
        }
#if HAVE_ASYNC_LDS
        asm volatile("s_wait_asynccnt 0" ::: "memory");
#endif
    } else {
        for (int t = threadIdx.x; t < 3 * WIN; t += blockDim.x) {
            int ry = t / WIN, xx = t - ry * WIN;
            int y = y0 - 1 + ry, x = xbase - 1 + xx;
            bool inb = (y >= 0 && y < H && x >= 0 && x < W);
            _Float16* dst = &sact[t * CPAD];
            const size_t pb = (size_t)n * hw + (size_t)y * W + x;
            for (int c = 0; c < CPAD; ++c) {
                _Float16 vv = (_Float16)0.f;
                if (inb && c < Ct)
                    vv = (c < C0) ? src0[pb * C0 + c] : src1[pb * C1 + (c - C0)];
                dst[c] = vv;
            }
        }
    }
    __syncthreads();

    const int hi  = lane >> 4;
    const int lx  = wave * 16 + (lane & 15);      // pixel within strip
    const int p   = y0 * W + xbase + lx;
    const int lxC = lx * CPAD;

    v8f acc = {};
#pragma unroll
    for (int ks = 0; ks < KSTEPS; ++ks) {
        v16h a = gA[(size_t)mtile * KSTEPS * 32 + ks * 32 + lane];
        // 16-K group -> compile-time LDS byte offsets after unroll
        const int g0 = ks * 2, g1 = ks * 2 + 1;
        const int r0 = g0 / CB, cb0 = g0 - r0 * CB;
        const int r1 = g1 / CB, cb1 = g1 - r1 * CB;
        const int off0 = ((r0 / 3) * WIN + (r0 % 3)) * CPAD + cb0 * 16;
        const int off1 = ((r1 / 3) * WIN + (r1 % 3)) * CPAD + cb1 * 16;
        const bool v1  = (g1 < GV);   // constant per step
        // tail hi-lanes re-load off0: their A-fragment K-range is packed zero
        int ofs = lxC + ((hi && v1) ? off1 : off0);
        v16h b = *(const v16h*)&sact[ofs];
        acc = __builtin_amdgcn_wmma_f32_16x16x32_f16(
                  false, a, false, b, (short)0, acc, false, false);
    }

    // D: lane -> pixel p; VGPR r -> m = hi*8 + r (m-contiguous run of 8)
    U8f du; du.v = acc;
    if (Cout == 16) {
        U8h o;
#pragma unroll
        for (int r = 0; r < 8; ++r) {
            int m = (hi << 3) + r;
            float vv = du.f[r] + (bias ? bias[m] : 0.f);
            if (relu) vv = fmaxf(vv, 0.f);
            o.h[r] = (_Float16)vv;
        }
        *(v8h*)(out + ((size_t)n * hw + p) * 16 + (hi << 3)) = o.v;
    } else {
#pragma unroll
        for (int r = 0; r < 8; ++r) {
            int m = mtile * 16 + (hi << 3) + r;
            if (m < Cout) {
                float vv = du.f[r] + (bias ? bias[m] : 0.f);
                if (relu) vv = fmaxf(vv, 0.f);
                out[((size_t)n * hw + p) * Cout + m] = (_Float16)vv;
            }
        }
    }
}

// ---------------------------------------------------------------------------
// Deformable 3x3 conv (mmcv style), src NHWC-f16 C=16, offsets NHWC-f16.
// KSTEPS fixed at 5 (K = 9*16 = 144, padded to 160); fully unrolled.
// Corner gathers are UNCONDITIONAL 16B v8h loads at clamped coordinates;
// out-of-bounds validity is folded into the bilinear weights (v_cndmask
// instead of EXEC branches).  Bilinear combine runs in packed f16.
// ---------------------------------------------------------------------------
__global__ __launch_bounds__(256)
void dconv3x3_nhwc_wmma_kernel(
    const _Float16* __restrict__ src,
    const _Float16* __restrict__ offs, int offC, int dg,
    int H, int W,
    const v16h* __restrict__ gA,
    int Cout, int relu,
    _Float16* __restrict__ outh, float* __restrict__ outf)
{
    constexpr int KSTEPS = 5;
    __shared__ __align__(32) v16h swgt[KSTEPS * 32];
    const int lane  = threadIdx.x & 31;
    const int wave  = threadIdx.x >> 5;
    const int mtile = blockIdx.y;
    const int n     = blockIdx.z;

    for (int i = threadIdx.x; i < KSTEPS * 32; i += 256)
        swgt[i] = gA[(size_t)mtile * KSTEPS * 32 + i];
    __syncthreads();

    const int hi = lane >> 4;
    const int pixtile = blockIdx.x * 8 + wave;
    const int p  = pixtile * 16 + (lane & 15);
    const int h0 = p / W, w0 = p - (p / W) * W;
    const size_t hw = (size_t)H * W;
    const _Float16* sbase = src + (size_t)n * hw * 16;
    const _Float16* ob    = offs + ((size_t)n * hw + p) * offC;
    const int Cg = 16 / dg;
    __builtin_prefetch(ob + offC * 16, 0, 1);     // next tile's offsets

    v8f acc = {};
#pragma unroll
    for (int ks = 0; ks < KSTEPS; ++ks) {
        v16h a = swgt[ks * 32 + lane];
        int kid = ks * 2 + hi;
        if (kid > 8) kid = 8;                     // tail clamp; A there is zero
        const int dy  = kid / 3 - 1;
        const int dxk = kid - (kid / 3) * 3 - 1;
        U16h bu;
#pragma unroll
        for (int h8 = 0; h8 < 2; ++h8) {
            int g = (h8 * 8) / Cg;                // deform group
            float offy = (float)ob[(g * 9 + kid) * 2 + 0];
            float offx = (float)ob[(g * 9 + kid) * 2 + 1];
            float py = offy + (float)(h0 + dy);
            float px = offx + (float)(w0 + dxk);
            float y0f = floorf(py), x0f = floorf(px);
            float wy = py - y0f, wx = px - x0f;
            int yi = (int)y0f, xi = (int)x0f;
            // clamped coords (always-valid addresses) + validity masks
            int yc0 = min(max(yi, 0), H - 1),     yc1 = min(max(yi + 1, 0), H - 1);
            int xc0 = min(max(xi, 0), W - 1),     xc1 = min(max(xi + 1, 0), W - 1);
            bool vy0 = (yi >= 0) & (yi < H),      vy1 = (yi + 1 >= 0) & (yi + 1 < H);
            bool vx0 = (xi >= 0) & (xi < W),      vx1 = (xi + 1 >= 0) & (xi + 1 < W);
            U8h u00, u01, u10, u11;
            u00.v = *(const v8h*)(sbase + ((size_t)yc0 * W + xc0) * 16 + h8 * 8);
            u01.v = *(const v8h*)(sbase + ((size_t)yc0 * W + xc1) * 16 + h8 * 8);
            u10.v = *(const v8h*)(sbase + ((size_t)yc1 * W + xc0) * 16 + h8 * 8);
            u11.v = *(const v8h*)(sbase + ((size_t)yc1 * W + xc1) * 16 + h8 * 8);
            _Float16 hw00 = (_Float16)((vy0 & vx0) ? (1.f - wy) * (1.f - wx) : 0.f);
            _Float16 hw01 = (_Float16)((vy0 & vx1) ? (1.f - wy) * wx         : 0.f);
            _Float16 hw10 = (_Float16)((vy1 & vx0) ? wy * (1.f - wx)         : 0.f);
            _Float16 hw11 = (_Float16)((vy1 & vx1) ? wy * wx                 : 0.f);
#pragma unroll
            for (int c = 0; c < 8; ++c)
                bu.h[h8 * 8 + c] = hw00 * u00.h[c] + hw01 * u01.h[c]
                                 + hw10 * u10.h[c] + hw11 * u11.h[c];
        }
        acc = __builtin_amdgcn_wmma_f32_16x16x32_f16(
                  false, a, false, bu.v, (short)0, acc, false, false);
    }

    U8f du; du.v = acc;
    if (outf) {                                   // final: f32 NCHW output
#pragma unroll
        for (int r = 0; r < 8; ++r) {
            int m = mtile * 16 + (hi << 3) + r;
            if (m < Cout) {
                float vv = du.f[r];
                if (relu) vv = fmaxf(vv, 0.f);
                outf[(size_t)(n * Cout + m) * hw + p] = vv;
            }
        }
    } else {                                      // f16 NHWC (Cout == 16)
        U8h o;
#pragma unroll
        for (int r = 0; r < 8; ++r) {
            float vv = du.f[r];
            if (relu) vv = fmaxf(vv, 0.f);
            o.h[r] = (_Float16)vv;
        }
        *(v8h*)(outh + ((size_t)n * hw + p) * 16 + (hi << 3)) = o.v;
    }
}

// --------------------------- MaxPool 3x3 s2 p1 (NHWC f16) ------------------
__global__ __launch_bounds__(256)
void maxpool_nhwc_kernel(const _Float16* __restrict__ in, _Float16* __restrict__ out,
                         int Nn, int C, int H, int W)
{
    const int Ho = H >> 1, Wo = W >> 1;
    long total = (long)Nn * Ho * Wo * C;
    long idx = (long)blockIdx.x * 256 + threadIdx.x;
    if (idx >= total) return;
    int c = (int)(idx % C);
    long t = idx / C;
    int wo = (int)(t % Wo); t /= Wo;
    int ho = (int)(t % Ho);
    int nn = (int)(t / Ho);
    float m = -3.402823466e38f;
    int hs = ho * 2 - 1, ws0 = wo * 2 - 1;
#pragma unroll
    for (int dy = 0; dy < 3; ++dy) {
        int y = hs + dy;
        if (y < 0 || y >= H) continue;
#pragma unroll
        for (int dx = 0; dx < 3; ++dx) {
            int x = ws0 + dx;
            if (x < 0 || x >= W) continue;
            m = fmaxf(m, (float)in[(((size_t)nn * H + y) * W + x) * C + c]);
        }
    }
    out[idx] = (_Float16)m;
}

// ----------------- Bilinear x2 upsample (half-pixel, clamped, NHWC f16) ----
__global__ __launch_bounds__(256)
void up2_nhwc_kernel(const _Float16* __restrict__ in, _Float16* __restrict__ out,
                     int Nn, int C, int H, int W)
{
    const int Ho = H * 2, Wo = W * 2;
    long total = (long)Nn * Ho * Wo * C;
    long idx = (long)blockIdx.x * 256 + threadIdx.x;
    if (idx >= total) return;
    int c = (int)(idx % C);
    long t = idx / C;
    int wo = (int)(t % Wo); t /= Wo;
    int ho = (int)(t % Ho);
    int nn = (int)(t / Ho);
    float sy = (ho + 0.5f) * 0.5f - 0.5f;
    float sx = (wo + 0.5f) * 0.5f - 0.5f;
    float y0f = floorf(sy), x0f = floorf(sx);
    float wy = sy - y0f, wx = sx - x0f;
    int y0 = (int)y0f, x0 = (int)x0f;
    int y0c = min(max(y0, 0), H - 1), y1c = min(max(y0 + 1, 0), H - 1);
    int x0c = min(max(x0, 0), W - 1), x1c = min(max(x0 + 1, 0), W - 1);
    const _Float16* s = in + (size_t)nn * H * W * C;
    float vv = (float)s[((size_t)y0c * W + x0c) * C + c] * (1.f - wy) * (1.f - wx)
             + (float)s[((size_t)y0c * W + x1c) * C + c] * (1.f - wy) * wx
             + (float)s[((size_t)y1c * W + x0c) * C + c] * wy * (1.f - wx)
             + (float)s[((size_t)y1c * W + x1c) * C + c] * wy * wx;
    out[idx] = (_Float16)vv;
}

// ---------------------------------------------------------------------------
// Host orchestration
// ---------------------------------------------------------------------------
struct PW { const v16h* p; int Ksteps, Cpad, Mtiles; };

extern "C" void kernel_launch(void* const* d_in, const int* in_sizes, int n_in,
                              void* d_out, int out_size, void* d_ws, size_t ws_size,
                              hipStream_t stream)
{
    (void)in_sizes; (void)n_in; (void)out_size; (void)ws_size;
    const int N = 4;

    const float* ref_img   = (const float*)d_in[0];
    const float* unreg_img = (const float*)d_in[1];
    const float* fe_w1[3] = {(const float*)d_in[2],  (const float*)d_in[6],  (const float*)d_in[10]};
    const float* fe_b1[3] = {(const float*)d_in[3],  (const float*)d_in[7],  (const float*)d_in[11]};
    const float* fe_w2[3] = {(const float*)d_in[4],  (const float*)d_in[8],  (const float*)d_in[12]};
    const float* fe_b2[3] = {(const float*)d_in[5],  (const float*)d_in[9],  (const float*)d_in[13]};
    const float* ob_w[3]  = {(const float*)d_in[14], (const float*)d_in[16], (const float*)d_in[18]};
    const float* ob_b[3]  = {(const float*)d_in[15], (const float*)d_in[17], (const float*)d_in[19]};
    const float* og_w[3]  = {(const float*)d_in[20], (const float*)d_in[22], (const float*)d_in[24]};
    const float* og_b[3]  = {(const float*)d_in[21], (const float*)d_in[23], (const float*)d_in[25]};
    const float* dc_w[3]  = {(const float*)d_in[26], (const float*)d_in[27], (const float*)d_in[28]};
    const float* fc_w[3]  = {(const float*)d_in[29], (const float*)d_in[31], (const float*)d_in[33]};
    const float* fc_b[3]  = {(const float*)d_in[30], (const float*)d_in[32], (const float*)d_in[34]};
    const float* dob_w = (const float*)d_in[35];
    const float* dob_b = (const float*)d_in[36];
    const float* dog_w = (const float*)d_in[37];
    const float* dog_b = (const float*)d_in[38];
    const float* ddc_w = (const float*)d_in[39];

    char* wsb = (char*)d_ws; size_t off = 0;
    auto alloc = [&](size_t bytes) -> void* {
        off = (off + 255) & ~(size_t)255;
        void* r = wsb + off; off += bytes; return r;
    };
    auto hbuf = [&](int C, int S) -> _Float16* {
        return (_Float16*)alloc((size_t)N * C * S * S * sizeof(_Float16));
    };

    auto pack = [&](const float* w, int Cout, int Cin) -> PW {
        PW r; r.Cpad = ((Cin + 15) / 16) * 16;
        r.Ksteps = (9 * r.Cpad + 31) / 32;
        r.Mtiles = (Cout + 15) / 16;
        size_t e = (size_t)r.Mtiles * r.Ksteps * 512;
        _Float16* pw = (_Float16*)alloc(e * sizeof(_Float16));
        pack_w_kernel<<<dim3((unsigned)((e + 255) / 256)), 256, 0, stream>>>(
            w, pw, Cout, Cin, r.Cpad, r.Ksteps, r.Mtiles);
        r.p = (const v16h*)pw;
        return r;
    };
    auto conv = [&](const _Float16* s0, int C0, const _Float16* s1, int C1,
                    const PW& pw, const float* bias, bool relu,
                    int S, int Cout, _Float16* outp) {
        int TS = S < 128 ? S : 128;
        int threads = (TS / 16) * 32;
        dim3 grid((unsigned)(S * (S / TS)), (unsigned)pw.Mtiles, (unsigned)N);
        int rl = (int)relu;
        if (TS == 128) {
            if      (pw.Cpad == 16) conv3x3_nhwc_wmma_kernel<16,128><<<grid, threads, 0, stream>>>(s0, C0, s1, C1, S, S, pw.p, bias, Cout, rl, outp);
            else if (pw.Cpad == 32) conv3x3_nhwc_wmma_kernel<32,128><<<grid, threads, 0, stream>>>(s0, C0, s1, C1, S, S, pw.p, bias, Cout, rl, outp);
            else                    conv3x3_nhwc_wmma_kernel<48,128><<<grid, threads, 0, stream>>>(s0, C0, s1, C1, S, S, pw.p, bias, Cout, rl, outp);
        } else {
            if      (pw.Cpad == 16) conv3x3_nhwc_wmma_kernel<16,64><<<grid, threads, 0, stream>>>(s0, C0, s1, C1, S, S, pw.p, bias, Cout, rl, outp);
            else if (pw.Cpad == 32) conv3x3_nhwc_wmma_kernel<32,64><<<grid, threads, 0, stream>>>(s0, C0, s1, C1, S, S, pw.p, bias, Cout, rl, outp);
            else                    conv3x3_nhwc_wmma_kernel<48,64><<<grid, threads, 0, stream>>>(s0, C0, s1, C1, S, S, pw.p, bias, Cout, rl, outp);
        }
    };
    auto dconv = [&](const _Float16* s, const _Float16* of, int offC, int dg,
                     const PW& pw, bool relu, int S, int Cout,
                     _Float16* outh, float* outf) {
        dim3 grid((unsigned)(S * S / 128), (unsigned)pw.Mtiles, (unsigned)N);
        dconv3x3_nhwc_wmma_kernel<<<grid, 256, 0, stream>>>(
            s, of, offC, dg, S, S, pw.p, Cout, (int)relu, outh, outf);
    };
    auto pool = [&](const _Float16* in, _Float16* outp, int C, int S) {
        long total = (long)N * (S / 2) * (S / 2) * C;
        maxpool_nhwc_kernel<<<dim3((unsigned)((total + 255) / 256)), 256, 0, stream>>>(
            in, outp, N, C, S, S);
    };
    auto up = [&](const _Float16* in, _Float16* outp, int C, int S) {  // S = in size
        long total = (long)N * (S * 2) * (S * 2) * C;
        up2_nhwc_kernel<<<dim3((unsigned)((total + 255) / 256)), 256, 0, stream>>>(
            in, outp, N, C, S, S);
    };

    // --- pack all weights ---
    PW p_feA[3] = {pack(fe_w1[0], 16, 1),  pack(fe_w1[1], 16, 16), pack(fe_w1[2], 16, 16)};
    PW p_feB[3] = {pack(fe_w2[0], 16, 16), pack(fe_w2[1], 16, 16), pack(fe_w2[2], 16, 16)};
    PW p_ob[3]  = {pack(ob_w[0], 16, 32),  pack(ob_w[1], 16, 32),  pack(ob_w[2], 16, 32)};
    PW p_og[3]  = {pack(og_w[0], 18, 34),  pack(og_w[1], 18, 34),  pack(og_w[2], 18, 16)};
    PW p_dc[3]  = {pack(dc_w[0], 16, 16),  pack(dc_w[1], 16, 16),  pack(dc_w[2], 16, 16)};
    PW p_fc[3]  = {pack(fc_w[0], 16, 32),  pack(fc_w[1], 16, 32),  pack(fc_w[2], 16, 16)};
    PW p_dob = pack(dob_w, 16, 32);
    PW p_dog = pack(dog_w, 36, 34);
    PW p_ddc = pack(ddc_w, 2, 16);

    // --- f16 copies of the input images (NCHW C=1 == NHWC C=1) ---
    long imgTot = (long)N * 512 * 512;
    _Float16* refh   = hbuf(1, 512);
    _Float16* unregh = hbuf(1, 512);
    f32_to_f16_kernel<<<dim3((unsigned)((imgTot + 255) / 256)), 256, 0, stream>>>(
        ref_img, refh, imgTot);
    f32_to_f16_kernel<<<dim3((unsigned)((imgTot + 255) / 256)), 256, 0, stream>>>(
        unreg_img, unregh, imgTot);

    // --- activation buffers (NHWC f16) ---
    _Float16* tA = hbuf(16, 512);
    _Float16* tB = hbuf(16, 512);
    _Float16* rf[3] = {hbuf(16, 256), hbuf(16, 128), hbuf(16, 64)};
    _Float16* uf[3] = {hbuf(16, 256), hbuf(16, 128), hbuf(16, 64)};
    _Float16* obuf  = hbuf(16, 256);
    _Float16* offb[3] = {hbuf(18, 256), hbuf(18, 128), hbuf(18, 64)};
    _Float16* upo  = hbuf(18, 256);
    _Float16* fb   = hbuf(16, 256);
    _Float16* upf  = hbuf(16, 256);
    _Float16* feat[3] = {hbuf(16, 256), hbuf(16, 128), hbuf(16, 64)};
    _Float16* do1 = hbuf(16, 256);
    _Float16* do2 = hbuf(36, 256);

    // --- feature pyramid for both images ---
    auto pyramid = [&](const _Float16* img, _Float16** f) {
        conv(img, 1, nullptr, 0, p_feA[0], fe_b1[0], true, 512, 16, tA);
        conv(tA, 16, nullptr, 0, p_feB[0], fe_b2[0], true, 512, 16, tB);
        pool(tB, f[0], 16, 512);
        conv(f[0], 16, nullptr, 0, p_feA[1], fe_b1[1], true, 256, 16, tA);
        conv(tA, 16, nullptr, 0, p_feB[1], fe_b2[1], true, 256, 16, tB);
        pool(tB, f[1], 16, 256);
        conv(f[1], 16, nullptr, 0, p_feA[2], fe_b1[2], true, 128, 16, tA);
        conv(tA, 16, nullptr, 0, p_feB[2], fe_b2[2], true, 128, 16, tB);
        pool(tB, f[2], 16, 128);
    };
    pyramid(refh, rf);
    pyramid(unregh, uf);

    // --- coarse-to-fine alignment ---
    const int S[3] = {256, 128, 64};
    for (int i = 2; i >= 0; --i) {
        conv(rf[i], 16, uf[i], 16, p_ob[i], ob_b[i], true, S[i], 16, obuf);
        if (i == 2) {
            conv(obuf, 16, nullptr, 0, p_og[2], og_b[2], true, 64, 18, offb[2]);
        } else {
            up(offb[i + 1], upo, 18, S[i + 1]);
            conv(obuf, 16, upo, 18, p_og[i], og_b[i], true, S[i], 18, offb[i]);
        }
        dconv(uf[i], offb[i], 18, 1, p_dc[i], true, S[i], 16, fb, nullptr);
        if (i == 2) {
            conv(fb, 16, nullptr, 0, p_fc[2], fc_b[2], true, 64, 16, feat[2]);
        } else {
            up(feat[i + 1], upf, 16, S[i + 1]);
            conv(fb, 16, upf, 16, p_fc[i], fc_b[i], true, S[i], 16, feat[i]);
        }
    }

    // --- final refinement + dg=2 deformable conv to 2-channel f32 output ---
    conv(rf[0], 16, feat[0], 16, p_dob, dob_b, false, 256, 16, do1);
    conv(do1, 16, offb[0], 18, p_dog, dog_b, false, 256, 36, do2);
    dconv(feat[0], do2, 36, 2, p_ddc, false, 256, 2, nullptr, (float*)d_out);
}
